// Model_88656714925250
// MI455X (gfx1250) — matrix operand order repaired and checked by reading, not compiled, not run on hardware
//
#include <hip/hip_runtime.h>
#include <hip/hip_bf16.h>

typedef __attribute__((ext_vector_type(16))) _Float16 v16h;
typedef __attribute__((ext_vector_type(8)))  _Float16 v8h;
typedef __attribute__((ext_vector_type(8)))  float    v8f;

#define BB   8
#define LL   512
#define CIN  55
#define DM   128
#define NH   8
#define EH   16
#define DFF  512
#define COUT 55
#define NLAY 3

__device__ __forceinline__ v16h loadA_frag(const _Float16* p) {
  const v8h lo = *(const v8h*)(p);
  const v8h hi = *(const v8h*)(p + 16);
  return __builtin_shufflevector(lo, hi, 0,1,2,3,4,5,6,7,8,9,10,11,12,13,14,15);
}

// ---------------------------------------------------------------------------
// FAST PATH: full tiles only (M%16==0, N%64==0, K%32==0), B pre-packed as
// [N/16][K][16] so each lane's fragment is one contiguous 32B load.
// One wave computes a 16x64 strip (4 accumulators). Software-pipelined:
// next K-step's fragments are loaded before the current step's 4 WMMAs.
// ---------------------------------------------------------------------------
__global__ void wmma_gemm_fast(
    const _Float16* __restrict__ A, int lda,
    const _Float16* __restrict__ Bp,
    const float* __restrict__ bias,
    float* __restrict__ C, int ldc,
    _Float16* __restrict__ Ch, int ldch,
    int M, int N, int K, int act)
{
  const int lane = threadIdx.x & 31;
  const int wave = threadIdx.x >> 5;
  const int stripsN = N >> 6;
  const int strip = blockIdx.x * (blockDim.x >> 5) + wave;
  if (strip >= (M >> 4) * stripsN) return;   // wave-uniform

  const int mt  = (strip / stripsN) << 4;
  const int nt4 = (strip % stripsN) << 2;    // first of 4 16-col tiles
  const int g = lane >> 4;
  const int r = lane & 15;

  const _Float16* aPtr  = A + (long)(mt + r) * lda + g * 8;
  const _Float16* bBase = Bp + ((long)nt4 * K + lane) * 16;
  const long bStride = (long)K * 16;         // halfs between 16-col tiles

  // Prologue: load K-chunk 0
  v16h af = loadA_frag(aPtr);
  v16h b0 = *(const v16h*)(bBase);
  v16h b1 = *(const v16h*)(bBase + bStride);
  v16h b2 = *(const v16h*)(bBase + 2 * bStride);
  v16h b3 = *(const v16h*)(bBase + 3 * bStride);

  v8f acc0 = {}, acc1 = {}, acc2 = {}, acc3 = {};
  for (int k0 = 32; k0 < K; k0 += 32) {
    // Issue next chunk's loads first (independent of current WMMAs)
    const _Float16* bp2 = bBase + (long)k0 * 16;
    const v16h afn = loadA_frag(aPtr + k0);
    const v16h b0n = *(const v16h*)(bp2);
    const v16h b1n = *(const v16h*)(bp2 + bStride);
    const v16h b2n = *(const v16h*)(bp2 + 2 * bStride);
    const v16h b3n = *(const v16h*)(bp2 + 3 * bStride);
    __builtin_prefetch(bp2 + 32 * 16, 0, 1);   // chunk k0+32 -> global_prefetch_b8
    acc0 = __builtin_amdgcn_wmma_f32_16x16x32_f16(false, af, false, b0, (short)0, acc0, false, false);
    acc1 = __builtin_amdgcn_wmma_f32_16x16x32_f16(false, af, false, b1, (short)0, acc1, false, false);
    acc2 = __builtin_amdgcn_wmma_f32_16x16x32_f16(false, af, false, b2, (short)0, acc2, false, false);
    acc3 = __builtin_amdgcn_wmma_f32_16x16x32_f16(false, af, false, b3, (short)0, acc3, false, false);
    af = afn; b0 = b0n; b1 = b1n; b2 = b2n; b3 = b3n;
  }
  // Epilogue: last chunk
  acc0 = __builtin_amdgcn_wmma_f32_16x16x32_f16(false, af, false, b0, (short)0, acc0, false, false);
  acc1 = __builtin_amdgcn_wmma_f32_16x16x32_f16(false, af, false, b1, (short)0, acc1, false, false);
  acc2 = __builtin_amdgcn_wmma_f32_16x16x32_f16(false, af, false, b2, (short)0, acc2, false, false);
  acc3 = __builtin_amdgcn_wmma_f32_16x16x32_f16(false, af, false, b3, (short)0, acc3, false, false);

  const v8f accs[4] = {acc0, acc1, acc2, acc3};
  const int rowBase = mt + (g << 3);
  #pragma unroll
  for (int t = 0; t < 4; ++t) {
    const int col = ((nt4 + t) << 4) + r;
    const float bv = bias ? bias[col] : 0.0f;
    #pragma unroll
    for (int j = 0; j < 8; ++j) {
      float v = accs[t][j] + bv;
      if (act == 1) v = 0.5f * v * (1.0f + erff(v * 0.7071067811865476f));
      if (C)  C[(long)(rowBase + j) * ldc + col] = v;
      if (Ch) Ch[(long)(rowBase + j) * ldch + col] = (_Float16)v;
    }
  }
}

// ---------------------------------------------------------------------------
// GENERIC (edge shapes): C[m,n] = act( sum_k A[m,k]*W[n,k] + bias[n] ),
// batched via blockIdx.y/z, bounds-checked, B row-major with k-stride wk.
// ---------------------------------------------------------------------------
__global__ void wmma_gemm(
    const _Float16* __restrict__ A, long aSY, long aSZ, int lda,
    const _Float16* __restrict__ W, long wSY, long wSZ, int ldw, int wk,
    const float* __restrict__ bias,
    float* __restrict__ C, long cSY, long cSZ, int ldc,
    _Float16* __restrict__ Ch, long chSY, long chSZ, int ldch,
    int M, int N, int K, int act)
{
  const int lane  = threadIdx.x & 31;
  const int wave  = threadIdx.x >> 5;
  const int tilesN = (N + 15) >> 4;
  const int tilesM = (M + 15) >> 4;
  const int tile = blockIdx.x * (blockDim.x >> 5) + wave;
  if (tile >= tilesM * tilesN) return;   // wave-uniform

  const long bY = blockIdx.y, bZ = blockIdx.z;
  A += bY * aSY + bZ * aSZ;
  W += bY * wSY + bZ * wSZ;
  if (C)  C  += bY * cSY  + bZ * cSZ;
  if (Ch) Ch += bY * chSY + bZ * chSZ;

  const int mt = (tile / tilesN) << 4;
  const int nt = (tile % tilesN) << 4;
  const int g  = lane >> 4;
  const int r  = lane & 15;

  const int  am   = mt + r;
  const bool aOk  = (am < M);
  const long aRow = (long)am * lda;
  const long wRow = (long)nt * ldw;
  const bool nFull = (nt + 16 <= N);

  v8f acc = {};
  for (int k0 = 0; k0 < K; k0 += 32) {
    v16h af = {};
    if (aOk) {
      const int kb = k0 + g * 8;
      #pragma unroll
      for (int j = 0; j < 16; ++j) {
        const int k = (j < 8) ? (kb + j) : (kb + 16 + (j - 8));
        if (k < K) af[j] = A[aRow + k];
      }
    }
    v16h wf = {};
    if (nt + r < N) {
      const int kb = k0 + g * 8;
      #pragma unroll
      for (int j = 0; j < 16; ++j) {
        const int k = (j < 8) ? (kb + j) : (kb + 16 + (j - 8));
        if (k < K) wf[j] = W[wRow + (long)r * ldw + (long)k * wk];
      }
    }
    acc = __builtin_amdgcn_wmma_f32_16x16x32_f16(false, af, false, wf,
                                                 (short)0, acc, false, false);
  }

  const int col = nt + r;
  if (col < N) {
    const float bv = bias ? bias[col] : 0.0f;
    #pragma unroll
    for (int j = 0; j < 8; ++j) {
      const int row = mt + (g << 3) + j;
      if (row < M) {
        float v = acc[j] + bv;
        if (act == 1) v = 0.5f * v * (1.0f + erff(v * 0.7071067811865476f));
        if (C)  C[(long)row * ldc + col] = v;
        if (Ch) Ch[(long)row * ldch + col] = (_Float16)v;
      }
    }
  }
}

// ---------------------------------------------------------------------------
__global__ void cvt_kernel(const float* __restrict__ src, _Float16* __restrict__ dst, long n) {
  long i = (long)blockIdx.x * blockDim.x + threadIdx.x;
  if (i < n) dst[i] = (_Float16)src[i];
}

// W [layers][N][K] f32 -> packed f16 [layers][N/16][K][16]
__global__ void pack_b_kernel(const float* __restrict__ src, _Float16* __restrict__ dst,
                              int K, long per, long total) {
  long i = (long)blockIdx.x * blockDim.x + threadIdx.x;
  if (i >= total) return;
  const long l  = i / per;
  const long rj = i % per;
  const int n = (int)(rj / K);
  const int k = (int)(rj % K);
  dst[l * per + ((long)(n >> 4) * K + (k & ~31) + ((k >> 3) & 1) * 16 + (n & 15)) * 16 + (k & 7) + ((k >> 4) & 1) * 8] = (_Float16)src[i];
}

// Token conv1d (k=3, circular) + sinusoidal PE -> h (f32 + f16 mirror)
__global__ void embed_kernel(const float* __restrict__ x, const float* __restrict__ tokw,
                             float* __restrict__ h32, _Float16* __restrict__ h16)
{
  const int d = threadIdx.x;            // 0..127
  const long bl = blockIdx.x;           // b*L + l
  const int l = (int)(bl & (LL - 1));
  const int b = (int)(bl >> 9);
  float acc = 0.f;
  #pragma unroll
  for (int j = 0; j < 3; ++j) {
    int ls = l + j - 1;
    if (ls < 0)   ls += LL;
    if (ls >= LL) ls -= LL;
    const float* xr = x + ((long)b * LL + ls) * CIN;
    const float* wr = tokw + (long)d * CIN * 3 + j;
    for (int c = 0; c < CIN; ++c) acc += xr[c] * wr[c * 3];
  }
  const int i2 = d >> 1;
  const float div = expf((float)(2 * i2) * (-9.210340371976184f / 128.0f));
  const float ang = (float)l * div;
  const float pe  = (d & 1) ? cosf(ang) : sinf(ang);
  const float v = acc + pe;
  h32[bl * DM + d] = v;
  h16[bl * DM + d] = (_Float16)v;
}

// Row softmax over 512 scores (scaled by 1/sqrt(16)); writes f32 + f16.
__global__ void softmax_kernel(const float* __restrict__ scores,
                               float* __restrict__ series,
                               _Float16* __restrict__ series16)
{
  const long row = blockIdx.x;
  const float* in = scores + row * LL;
  float*    o32 = series   + row * LL;
  _Float16* o16 = series16 + row * LL;
  __shared__ float sh[128];
  const int t = threadIdx.x;
  float vals[4];
  float m = -3.4e38f;
  #pragma unroll
  for (int i = 0; i < 4; ++i) { vals[i] = in[t + i * 128] * 0.25f; m = fmaxf(m, vals[i]); }
  sh[t] = m; __syncthreads();
  for (int off = 64; off > 0; off >>= 1) { if (t < off) sh[t] = fmaxf(sh[t], sh[t + off]); __syncthreads(); }
  const float rm = sh[0]; __syncthreads();
  float sum = 0.f;
  #pragma unroll
  for (int i = 0; i < 4; ++i) { vals[i] = expf(vals[i] - rm); sum += vals[i]; }
  sh[t] = sum; __syncthreads();
  for (int off = 64; off > 0; off >>= 1) { if (t < off) sh[t] += sh[t + off]; __syncthreads(); }
  const float inv = 1.0f / sh[0];
  #pragma unroll
  for (int i = 0; i < 4; ++i) {
    const float p = vals[i] * inv;
    o32[t + i * 128] = p;
    o16[t + i * 128] = (_Float16)p;
  }
}

// Gaussian prior from raw sigma projection sraw[B*L, H].
__global__ void prior_kernel(const float* __restrict__ sraw, float* __restrict__ prior)
{
  const int row = blockIdx.x;              // ((b*H + h)*L + l)
  const int l  = row & (LL - 1);
  const int bh = row >> 9;
  const int hh = bh & (NH - 1);
  const int b  = bh >> 3;
  const float s  = sraw[((long)b * LL + l) * NH + hh];
  const float sg = 1.0f / (1.0f + expf(-5.0f * s)) + 1e-5f;
  const float sig = exp2f(sg * 1.5849625007211562f) - 1.0f;   // 3^sg - 1
  const float ninv = -1.0f / (2.0f * sig * sig);
  const float coef = 0.3989422804014327f / sig;
  float* out = prior + (long)row * LL;
  for (int sI = threadIdx.x; sI < LL; sI += blockDim.x) {
    const float d = (float)(l - sI);
    out[sI] = coef * expf(d * d * ninv);
  }
}

// h = LayerNorm(h + delta) * g + b  (in place; delta may be null); f32 + f16.
__global__ void add_ln_kernel(float* __restrict__ h32, _Float16* __restrict__ h16,
                              const float* __restrict__ delta,
                              const float* __restrict__ g, const float* __restrict__ bt)
{
  const int d = threadIdx.x;     // 128 = D
  const long row = blockIdx.x;
  const long idx = row * DM + d;
  float v = h32[idx] + (delta ? delta[idx] : 0.0f);
  __shared__ float sh[128];
  sh[d] = v; __syncthreads();
  for (int off = 64; off > 0; off >>= 1) { if (d < off) sh[d] += sh[d + off]; __syncthreads(); }
  const float mean = sh[0] * (1.0f / DM);
  __syncthreads();
  const float c = v - mean;
  sh[d] = c * c; __syncthreads();
  for (int off = 64; off > 0; off >>= 1) { if (d < off) sh[d] += sh[d + off]; __syncthreads(); }
  const float var = sh[0] * (1.0f / DM);
  const float o = c * rsqrtf(var + 1e-5f) * g[d] + bt[d];
  h32[idx] = o;
  h16[idx] = (_Float16)o;
}

// ---------------------------------------------------------------------------
static inline void* ws_take(char*& p, size_t bytes) {
  void* r = (void*)p;
  p += (bytes + 255) & ~(size_t)255;
  return r;
}

extern "C" void kernel_launch(void* const* d_in, const int* in_sizes, int n_in,
                              void* d_out, int out_size, void* d_ws, size_t ws_size,
                              hipStream_t stream) {
  (void)in_sizes; (void)n_in; (void)out_size; (void)ws_size;
  const float* x    = (const float*)d_in[0];
  const float* tokw = (const float*)d_in[1];
  const float* Wq   = (const float*)d_in[2];
  const float* bq   = (const float*)d_in[3];
  const float* Wk   = (const float*)d_in[4];
  const float* bk   = (const float*)d_in[5];
  const float* Wv   = (const float*)d_in[6];
  const float* bv   = (const float*)d_in[7];
  const float* Wsg  = (const float*)d_in[8];
  const float* bs   = (const float*)d_in[9];
  const float* Wo   = (const float*)d_in[10];
  const float* bo   = (const float*)d_in[11];
  const float* c1w  = (const float*)d_in[12];
  const float* c1b  = (const float*)d_in[13];
  const float* c2w  = (const float*)d_in[14];
  const float* c2b  = (const float*)d_in[15];
  const float* ln1g = (const float*)d_in[16];
  const float* ln1b = (const float*)d_in[17];
  const float* ln2g = (const float*)d_in[18];
  const float* ln2b = (const float*)d_in[19];
  const float* ng   = (const float*)d_in[20];
  const float* nb   = (const float*)d_in[21];
  const float* pw   = (const float*)d_in[22];
  const float* pb   = (const float*)d_in[23];

  float* out        = (float*)d_out;
  float* series_out = out + (long)BB * LL * COUT;                    // 225280
  float* prior_out  = series_out + (long)NLAY * BB * NH * LL * LL;   // +50331648

  // Workspace carve-up
  char* wsp = (char*)d_ws;
  _Float16* wqP   = (_Float16*)ws_take(wsp, (size_t)NLAY * DM * DM * 2);   // packed
  _Float16* wkP   = (_Float16*)ws_take(wsp, (size_t)NLAY * DM * DM * 2);
  _Float16* wvP   = (_Float16*)ws_take(wsp, (size_t)NLAY * DM * DM * 2);
  _Float16* woP   = (_Float16*)ws_take(wsp, (size_t)NLAY * DM * DM * 2);
  _Float16* c1P   = (_Float16*)ws_take(wsp, (size_t)NLAY * DFF * DM * 2);
  _Float16* c2P   = (_Float16*)ws_take(wsp, (size_t)NLAY * DM * DFF * 2);
  _Float16* ws16  = (_Float16*)ws_take(wsp, (size_t)NLAY * NH * DM * 2);   // row-major
  _Float16* pw16  = (_Float16*)ws_take(wsp, (size_t)COUT * DM * 2);        // row-major
  float*    h32   = (float*)   ws_take(wsp, (size_t)BB * LL * DM * 4);
  _Float16* h16   = (_Float16*)ws_take(wsp, (size_t)BB * LL * DM * 2);
  _Float16* q16   = (_Float16*)ws_take(wsp, (size_t)BB * LL * DM * 2);
  _Float16* k16   = (_Float16*)ws_take(wsp, (size_t)BB * LL * DM * 2);
  _Float16* v16   = (_Float16*)ws_take(wsp, (size_t)BB * LL * DM * 2);
  _Float16* at16  = (_Float16*)ws_take(wsp, (size_t)BB * LL * DM * 2);
  float*    sraw  = (float*)   ws_take(wsp, (size_t)BB * LL * NH * 4);
  float*    scores= (float*)   ws_take(wsp, (size_t)BB * NH * LL * LL * 4);   // 64 MB
  _Float16* ser16 = (_Float16*)ws_take(wsp, (size_t)BB * NH * LL * LL * 2);   // 32 MB
  float*    o32   = (float*)   ws_take(wsp, (size_t)BB * LL * DM * 4);
  _Float16* ff116 = (_Float16*)ws_take(wsp, (size_t)BB * LL * DFF * 2);
  float*    ff232 = (float*)   ws_take(wsp, (size_t)BB * LL * DM * 4);

  auto cvt = [&](const float* s, _Float16* dd, long n) {
    cvt_kernel<<<dim3((unsigned)((n + 255) / 256)), dim3(256), 0, stream>>>(s, dd, n);
  };
  auto pack = [&](const float* s, _Float16* dd, int N, int K, int layers) {
    const long per = (long)N * K;
    const long tot = per * layers;
    pack_b_kernel<<<dim3((unsigned)((tot + 255) / 256)), dim3(256), 0, stream>>>(s, dd, K, per, tot);
  };
  auto gemmF = [&](const _Float16* A, int lda, const _Float16* Bp, const float* bias,
                   float* C, int ldc, _Float16* Ch, int ldch, int M, int N, int K, int act) {
    const int waves = (M >> 4) * (N >> 6);
    wmma_gemm_fast<<<dim3((unsigned)((waves + 3) / 4)), dim3(128), 0, stream>>>(
        A, lda, Bp, bias, C, ldc, Ch, ldch, M, N, K, act);
  };
  auto gemm = [&](const _Float16* A, long aSY, long aSZ, int lda,
                  const _Float16* Wm, long wSY, long wSZ, int ldw, int wk,
                  const float* bias,
                  float* C, long cSY, long cSZ, int ldc,
                  _Float16* Ch, long chSY, long chSZ, int ldch,
                  int M, int N, int K, int act, int by, int bz) {
    const int tiles = ((M + 15) / 16) * ((N + 15) / 16);
    dim3 grid((unsigned)((tiles + 3) / 4), (unsigned)by, (unsigned)bz);
    wmma_gemm<<<grid, dim3(128), 0, stream>>>(A, aSY, aSZ, lda, Wm, wSY, wSZ, ldw, wk,
                                              bias, C, cSY, cSZ, ldc, Ch, chSY, chSZ, ldch,
                                              M, N, K, act);
  };

  // One-time weight preparation (deterministic each call)
  pack(Wq,  wqP, DM,  DM,  NLAY);
  pack(Wk,  wkP, DM,  DM,  NLAY);
  pack(Wv,  wvP, DM,  DM,  NLAY);
  pack(Wo,  woP, DM,  DM,  NLAY);
  pack(c1w, c1P, DFF, DM,  NLAY);
  pack(c2w, c2P, DM,  DFF, NLAY);
  cvt(Wsg, ws16, (long)NLAY * NH * DM);
  cvt(pw,  pw16, (long)COUT * DM);

  // Embedding
  embed_kernel<<<dim3(BB * LL), dim3(DM), 0, stream>>>(x, tokw, h32, h16);

  const int M4096 = BB * LL;
  for (int i = 0; i < NLAY; ++i) {
    // Q, K, V projections (fast path, f16 outputs)
    gemmF(h16, DM, wqP + (long)i * DM * DM, bq + i * DM, nullptr, 0, q16, DM, M4096, DM, DM, 0);
    gemmF(h16, DM, wkP + (long)i * DM * DM, bk + i * DM, nullptr, 0, k16, DM, M4096, DM, DM, 0);
    gemmF(h16, DM, wvP + (long)i * DM * DM, bv + i * DM, nullptr, 0, v16, DM, M4096, DM, DM, 0);
    // sigma projection (generic: N=8)
    gemm(h16,0,0,DM, ws16 + (long)i * NH * DM,0,0,DM,1, bs + i*NH,
         sraw,0,0,NH, nullptr,0,0,0, M4096,NH,DM,0, 1,1);

    // scores[b,h] = Q_h @ K_h^T  (generic: K=16; batch y=h, z=b)
    gemm(q16,(long)EH,(long)LL*DM,DM, k16,(long)EH,(long)LL*DM,DM,1, nullptr,
         scores,(long)LL*LL,(long)NH*LL*LL,LL, nullptr,0,0,0, LL,LL,EH,0, NH,BB);

    // softmax -> series (d_out) + f16 copy; prior -> d_out
    softmax_kernel<<<dim3(BB * NH * LL), dim3(128), 0, stream>>>(
        scores, series_out + (long)i * BB * NH * LL * LL, ser16);
    prior_kernel<<<dim3(BB * NH * LL), dim3(256), 0, stream>>>(
        sraw, prior_out + (long)i * BB * NH * LL * LL);

    // O[b,h] = series @ V_h  (generic: N=16, B transposed via wk=DM)
    gemm(ser16,(long)LL*LL,(long)NH*LL*LL,LL, v16,(long)EH,(long)LL*DM,1,DM, nullptr,
         nullptr,0,0,0, at16,(long)EH,(long)LL*DM,DM, LL,EH,LL,0, NH,BB);

    // output projection + residual LN1 (fast path)
    gemmF(at16, DM, woP + (long)i * DM * DM, bo + i * DM, o32, DM, nullptr, 0, M4096, DM, DM, 0);
    add_ln_kernel<<<dim3(M4096), dim3(DM), 0, stream>>>(h32, h16, o32, ln1g + i*DM, ln1b + i*DM);

    // FFN (fast path): gelu(h @ c1w^T + c1b) @ c2w^T + c2b, residual LN2
    gemmF(h16,   DM,  c1P + (long)i * DFF * DM, c1b + i * DFF, nullptr, 0, ff116, DFF, M4096, DFF, DM, 1);
    gemmF(ff116, DFF, c2P + (long)i * DM * DFF, c2b + i * DM,  ff232, DM, nullptr, 0,  M4096, DM, DFF, 0);
    add_ln_kernel<<<dim3(M4096), dim3(DM), 0, stream>>>(h32, h16, ff232, ln2g + i*DM, ln2b + i*DM);
  }

  // Final norm + projection to C_OUT (generic: N=55)
  add_ln_kernel<<<dim3(M4096), dim3(DM), 0, stream>>>(h32, h16, nullptr, ng, nb);
  gemm(h16,0,0,DM, pw16,0,0,DM,1, pb, out,0,0,COUT, nullptr,0,0,0, M4096,COUT,DM,0, 1,1);
}
